// AVFusion_83519934038576
// MI455X (gfx1250) — matrix operand order repaired and checked by reading, not compiled, not run on hardware
//
#include <hip/hip_runtime.h>

// ---------------------------------------------------------------------------
// CDNA5 (gfx1250, wave32) bf16 WMMA implementation of the AV-fusion block.
// Wave tile 64x64 (4x4 WMMA tiles); K = N = 1024 compile-time so all fragment
// loads use two base pointers + 24-bit immediate offsets (no in-loop VALU
// address math -> no WMMA->VALU hazard NOPs).
// ---------------------------------------------------------------------------

typedef __attribute__((ext_vector_type(16))) __bf16 v16bf;
typedef __attribute__((ext_vector_type(8)))  float  v8f;

union Frag {
    v16bf v;
    uint4 q[2];   // 32 bytes
};

// ---------------------------------------------------------------------------
// fp32 -> bf16 conversion (grid-stride)
// ---------------------------------------------------------------------------
__global__ void cvt_f32_bf16(const float* __restrict__ in,
                             __bf16* __restrict__ out, int n) {
    int i = blockIdx.x * blockDim.x + threadIdx.x;
    int stride = gridDim.x * blockDim.x;
    for (; i < n; i += stride) out[i] = (__bf16)in[i];
}

// ---------------------------------------------------------------------------
// C[M,1024] = act( X[M,1024] @ W[1024,1024]^T + bias )
//   X, W   : bf16, K-contiguous rows, K = N = 1024 fixed
//   outB   : optional bf16 result ; outF : optional f32 result
// Block tile 256(M) x 128(N), 256 threads = 8 waves in 4(M) x 2(N) grid,
// wave tile 64x64 = 4x4 WMMA tiles, K-step 32.  Requires M % 256 == 0.
// ---------------------------------------------------------------------------
__global__ __launch_bounds__(256) void gemm_bf16_wmma(
    const __bf16* __restrict__ X, const __bf16* __restrict__ W,
    const float* __restrict__ bias,
    __bf16* __restrict__ outB, float* __restrict__ outF,
    int M, int relu)
{
    constexpr int K = 1024;
    constexpr int N = 1024;

    const int lane  = threadIdx.x & 31;
    const int w     = threadIdx.x >> 5;
    const int waveM = w >> 1;          // 0..3
    const int waveN = w & 1;           // 0..1
    const int R  = blockIdx.y * 256 + waveM * 64;   // wave row base (64 rows)
    const int Cb = blockIdx.x * 128 + waveN * 64;   // wave col base (64 cols)
    const int mr = lane & 15;
    const int hi = (lane >> 4) & 1;

    // Per-lane base pointers; everything else is a compile-time immediate.
    // A fragment (16x32 bf16): lane<16 holds K {0..7, 16..23} of row mr,
    // lane>=16 holds K {8..15, 24..31}  (ISA 7.12.2)
    const __bf16* pA = X + ((size_t)(R + mr) << 10) + (hi ? 8 : 0);
    // B fragment (32x16 bf16): lane<16 holds K {0..7, 16..23} of column mr,
    // lane>=16 holds K {8..15, 24..31}. W is stored [N][K] so K is contiguous.
    const __bf16* pB = W + ((size_t)(Cb + mr) << 10) + (hi ? 8 : 0);

    v8f acc[4][4];
#pragma unroll
    for (int i = 0; i < 4; ++i)
#pragma unroll
        for (int j = 0; j < 4; ++j)
#pragma unroll
            for (int e = 0; e < 8; ++e) acc[i][j][e] = 0.0f;

    for (int k0 = 0; k0 < K; k0 += 32) {
        Frag a[4], b[4];
#pragma unroll
        for (int rt = 0; rt < 4; ++rt) {
            a[rt].q[0] = *reinterpret_cast<const uint4*>(pA + rt * 16 * K);
            a[rt].q[1] = *reinterpret_cast<const uint4*>(pA + rt * 16 * K + 16);
        }
#pragma unroll
        for (int ct = 0; ct < 4; ++ct) {
            b[ct].q[0] = *reinterpret_cast<const uint4*>(pB + ct * 16 * K);
            b[ct].q[1] = *reinterpret_cast<const uint4*>(pB + ct * 16 * K + 16);
        }
#pragma unroll
        for (int rt = 0; rt < 4; ++rt)
#pragma unroll
            for (int ct = 0; ct < 4; ++ct)
                acc[rt][ct] = __builtin_amdgcn_wmma_f32_16x16x32_bf16(
                    false, a[rt].v, false, b[ct].v,
                    (short)0, acc[rt][ct], false, false);
        pA += 32;
        pB += 32;
    }

    // Epilogue: bias + optional ReLU, store.
    // D layout: VGPR j -> row (j + hi*8), col = lane&15  (ISA 7.12.2)
#pragma unroll
    for (int ct = 0; ct < 4; ++ct) {
        const int col = Cb + ct * 16 + mr;
        const float bb = bias ? bias[col] : 0.0f;
#pragma unroll
        for (int rt = 0; rt < 4; ++rt) {
#pragma unroll
            for (int j = 0; j < 8; ++j) {
                const int row = R + rt * 16 + j + (hi ? 8 : 0);
                float vv = acc[rt][ct][j] + bb;
                if (relu) vv = fmaxf(vv, 0.0f);
                const size_t idx = ((size_t)row << 10) + col;
                if (outB) outB[idx] = (__bf16)vv;
                if (outF) outF[idx] = vv;
            }
        }
    }
    (void)M;
}

// ---------------------------------------------------------------------------
// Attention over the size-2 A/V axis.
// One block per output row r = (b*32+s)*64+g ; wave = head ; lane covers 4 dims.
// ---------------------------------------------------------------------------
__global__ __launch_bounds__(256) void attn_av(
    const __bf16* __restrict__ q,  const __bf16* __restrict__ kA,
    const __bf16* __restrict__ kV, const __bf16* __restrict__ vA,
    const __bf16* __restrict__ vV, __bf16* __restrict__ out)
{
    const int r = blockIdx.x;              // 0..32767
    const int b = r >> 11;                 // / (32*64)
    const int s = (r >> 6) & 31;
    const int g = r & 63;
    const int h = threadIdx.x >> 5;        // head 0..7
    const int lane = threadIdx.x & 31;

    const size_t qoff = ((size_t)(b * 32 + s)) * 1024 + h * 128 + lane * 4;
    const size_t koff = ((size_t)(b * 64 + g)) * 1024 + h * 128 + lane * 4;

    float qv[4], ka4[4], kv4[4], va4[4], vv4[4];
#pragma unroll
    for (int i = 0; i < 4; ++i) {
        qv[i]  = (float)q [qoff + i];
        ka4[i] = (float)kA[koff + i];
        kv4[i] = (float)kV[koff + i];
        va4[i] = (float)vA[koff + i];
        vv4[i] = (float)vV[koff + i];
    }
    float pa = 0.0f, pv = 0.0f;
#pragma unroll
    for (int i = 0; i < 4; ++i) { pa += qv[i] * ka4[i]; pv += qv[i] * kv4[i]; }
#pragma unroll
    for (int off = 16; off > 0; off >>= 1) {
        pa += __shfl_xor(pa, off, 32);
        pv += __shfl_xor(pv, off, 32);
    }
    const float scale = 0.08838834764831845f;  // 1/sqrt(128)
    const float sa = pa * scale, sv = pv * scale;
    const float m  = fmaxf(sa, sv);
    const float ea = __expf(sa - m), ev = __expf(sv - m);
    const float inv = 1.0f / (ea + ev);
    const float wa = ea * inv, wv = ev * inv;

    const size_t ooff = (size_t)r * 1024 + h * 128 + lane * 4;
#pragma unroll
    for (int i = 0; i < 4; ++i)
        out[ooff + i] = (__bf16)(wa * va4[i] + wv * vv4[i]);
}

// ---------------------------------------------------------------------------
// Host-side orchestration
// ---------------------------------------------------------------------------
extern "C" void kernel_launch(void* const* d_in, const int* in_sizes, int n_in,
                              void* d_out, int out_size, void* d_ws, size_t ws_size,
                              hipStream_t stream) {
    const float* A  = (const float*)d_in[0];
    const float* V  = (const float*)d_in[1];
    const float* S  = (const float*)d_in[2];
    const float* wA = (const float*)d_in[3];  const float* bA = (const float*)d_in[4];
    const float* wV = (const float*)d_in[5];  const float* bV = (const float*)d_in[6];
    const float* wS = (const float*)d_in[7];  const float* bS = (const float*)d_in[8];
    const float* wq = (const float*)d_in[9];  const float* bq = (const float*)d_in[10];
    const float* wk = (const float*)d_in[11]; const float* bk = (const float*)d_in[12];
    const float* wv = (const float*)d_in[13]; const float* bv = (const float*)d_in[14];
    const float* w1 = (const float*)d_in[15]; const float* b1 = (const float*)d_in[16];
    const float* w2 = (const float*)d_in[17]; const float* b2 = (const float*)d_in[18];
    float* out = (float*)d_out;

    const size_t EA = (size_t)16 * 64 * 1024;        // 1,048,576 (A/V rows)
    const size_t ES = (size_t)16 * 32 * 1024;        //   524,288 (S rows)
    const size_t EW = (size_t)1024 * 1024;           // 1,048,576 (weights)
    const size_t EO = (size_t)16 * 32 * 64 * 1024;   // 33,554,432 (FF rows)

    char* p = (char*)d_ws;
    auto take = [&](size_t elems) {
        __bf16* r = (__bf16*)p;
        p += elems * sizeof(__bf16);
        return r;
    };
    __bf16* bfA  = take(EA);
    __bf16* bfV  = take(EA);
    __bf16* bfS  = take(ES);
    __bf16* bwA  = take(EW);
    __bf16* bwV  = take(EW);
    __bf16* bwS  = take(EW);
    __bf16* bwq  = take(EW);
    __bf16* bwk  = take(EW);
    __bf16* bwv  = take(EW);
    __bf16* bw1  = take(EW);
    __bf16* bw2  = take(EW);
    __bf16* X1   = take(EA);   // A projection  (1024 x 1024)  -- adjacent to X2
    __bf16* X2   = take(EA);   // V projection
    __bf16* X3   = take(ES);   // S projection  (512 x 1024)
    __bf16* qb   = take(ES);
    __bf16* kAb  = take(EW);   // adjacent to kVb
    __bf16* kVb  = take(EW);
    __bf16* vAb  = take(EW);   // adjacent to vVb
    __bf16* vVb  = take(EW);
    __bf16* attn = take(EO);   // (32768 x 1024)
    __bf16* hid  = take(EO);
    (void)ws_size; (void)n_in; (void)in_sizes; (void)out_size;

    auto cvt = [&](const float* src, __bf16* dst, size_t n) {
        int blocks = (int)((n + 255) / 256);
        cvt_f32_bf16<<<blocks, 256, 0, stream>>>(src, dst, (int)n);
    };
    cvt(A,  bfA, EA);  cvt(V,  bfV, EA);  cvt(S,  bfS, ES);
    cvt(wA, bwA, EW);  cvt(wV, bwV, EW);  cvt(wS, bwS, EW);
    cvt(wq, bwq, EW);  cvt(wk, bwk, EW);  cvt(wv, bwv, EW);
    cvt(w1, bw1, EW);  cvt(w2, bw2, EW);

    auto gemm = [&](const __bf16* Xp, const __bf16* Wp, const float* bias,
                    __bf16* oB, float* oF, int M, int relu) {
        dim3 grid(1024 / 128, M / 256);
        gemm_bf16_wmma<<<grid, 256, 0, stream>>>(Xp, Wp, bias, oB, oF, M, relu);
    };

    // Input projections
    gemm(bfA, bwA, bA, X1, nullptr, 1024, 0);
    gemm(bfV, bwV, bV, X2, nullptr, 1024, 0);
    gemm(bfS, bwS, bS, X3, nullptr,  512, 0);
    // q projection
    gemm(X3, bwq, bq, qb, nullptr, 512, 0);
    // k / v projections: [X1;X2] is contiguous (2048 rows), outputs [kA;kV]
    // and [vA;vV] are contiguous too -> one launch each.
    gemm(X1, bwk, bk, kAb, nullptr, 2048, 0);
    gemm(X1, bwv, bv, vAb, nullptr, 2048, 0);
    // Attention over the A/V axis (size 2)
    attn_av<<<32768, 256, 0, stream>>>(qb, kAb, kVb, vAb, vVb, attn);
    // Position-wise FF: relu(attn @ w1^T + b1) @ w2^T + b2  -> d_out (fp32)
    gemm(attn, bw1, b1, hid, nullptr, 32768, 1);
    gemm(hid,  bw2, b2, nullptr, out, 32768, 0);
}
